// VectorQuantizerEMA_14585708937919
// MI455X (gfx1250) — matrix operand order repaired and checked by reading, not compiled, not run on hardware
//
#include <hip/hip_runtime.h>
#include <hip/hip_bf16.h>
#include <math.h>

// ---- problem constants -----------------------------------------------------
#define BB      32
#define DD      64
#define TPP     2048
#define KCODES  1024
#define NROWS   (BB * TPP)           // 65536
#define ZQ_SIZE (BB * DD * TPP)      // 4194304
#define TOK_OFF ZQ_SIZE              // token_ids start
#define CL_OFF  (ZQ_SIZE + NROWS)    // commit loss
#define PP_OFF  (CL_OFF + 1)         // perplexity

// ---- CDNA5 WMMA types ------------------------------------------------------
typedef __attribute__((ext_vector_type(16))) __bf16 v16bf;
typedef __attribute__((ext_vector_type(8)))  float  v8f;

union BF16x16 { v16bf v; unsigned short u[16]; };

__device__ __forceinline__ unsigned short f2bf(float f) {
  unsigned int u = __float_as_uint(f);
  unsigned int r = u + 0x7FFFu + ((u >> 16) & 1u);   // round-to-nearest-even
  return (unsigned short)(r >> 16);
}
__device__ __forceinline__ float bf2f(unsigned short h) {
  return __uint_as_float(((unsigned int)h) << 16);
}

// ---- workspace layout ------------------------------------------------------
// [0,4096)        : hist u32[1024]
// [4096,4104)     : loss accumulator (double)
// [8192,12288)    : cnorm f32[1024]
// [12288,+128KB)  : packed codebook hi (bf16, B-matrix lane layout)
// [143360,+128KB) : packed codebook lo

// ============================================================================
__global__ void vq_zero(unsigned int* __restrict__ hist, double* __restrict__ loss) {
  int t = threadIdx.x;
  if (t < KCODES) hist[t] = 0u;
  if (t == 0) *loss = 0.0;
}

// Pack codebook into CDNA5 B-matrix (32x16 bf16 K-slab) lane layout and
// compute ||c||^2.  Layout: [tile ct(64)][slab(2)][lane(32)][halfs(16)],
// lane = (d_in_slab>=16 ? 16:0) + (k&15), half j = d_in_slab&15.
__global__ void vq_prep(const float* __restrict__ cb,
                        unsigned short* __restrict__ cb_hi,
                        unsigned short* __restrict__ cb_lo,
                        float* __restrict__ cnorm) {
  int k = blockIdx.x * blockDim.x + threadIdx.x;
  if (k >= KCODES) return;
  const float* row = cb + k * DD;
  int ct = k >> 4, ln = k & 15;
  float s = 0.f;
  for (int d = 0; d < DD; ++d) {
    float c = row[d];
    s += c * c;
    int slab = d >> 5, dd = d & 31;
    int lane = ((dd >> 4) << 4) + ln;
    int off  = (((ct << 1) + slab) * 32 + lane) * 16 + (dd & 15);
    unsigned short hi = f2bf(c);
    cb_hi[off] = hi;
    cb_lo[off] = f2bf(c - bf2f(hi));
  }
  cnorm[k] = s;
}

// ============================================================================
// Main kernel: 512 blocks x 256 threads (8 waves).  Wave w owns 16 rows of
// ze_flat; each chunk iteration stages 64 codes (hi+lo bf16) in LDS and runs
// 4 N-tiles x 2 K-slabs x 3 WMMAs (bf16x3 compensated fp32 emulation).
// dist(m,n) = ||c_n||^2 + (-2 x_m)·c_n   (row-constant ||x||^2 dropped).
__launch_bounds__(256)
__global__ void vq_argmin(const float* __restrict__ ze,
                          const unsigned short* __restrict__ cb_hi,
                          const unsigned short* __restrict__ cb_lo,
                          const float* __restrict__ cnorm,
                          float* __restrict__ out_tok_f,
                          unsigned int* __restrict__ hist) {
  __shared__ uint4 lds_hi[512];   // 4096 bf16 = 8 KB
  __shared__ uint4 lds_lo[512];   // 8 KB

  const int tid  = threadIdx.x;
  const int wave = tid >> 5;
  const int lane = tid & 31;
  const int m    = lane & 15;     // row within tile (A) / column (B,C)
  const int h    = lane >> 4;     // lane-half
  const int rowbase = blockIdx.x * 128 + wave * 16;
  const int n = rowbase + m;
  const int b = n >> 11;          // TP = 2048
  const int t = n & 2047;
  const float* zb = ze + (size_t)b * DD * TPP + t;   // element d at zb[d*TPP]

  // Build A operands (-2x, split hi/lo) in a 16x32 bf16 A layout (contiguous k, same as the B packing):
  // element e -> K = h*16 + e within the slab.
  BF16x16 ah[2], al[2];
  #pragma unroll
  for (int s = 0; s < 2; ++s) {
    #pragma unroll
    for (int e = 0; e < 16; ++e) {
      int d = s * 32 + h * 16 + e;
      float x = -2.0f * zb[(size_t)d * TPP];
      unsigned short hi = f2bf(x);
      ah[s].u[e] = hi;
      al[s].u[e] = f2bf(x - bf2f(hi));
    }
  }

  float rv[8];
  int   ri[8];
  #pragma unroll
  for (int r = 0; r < 8; ++r) { rv[r] = 3.4e38f; ri[r] = 0; }

  const v16bf* Bh = (const v16bf*)lds_hi;
  const v16bf* Bl = (const v16bf*)lds_lo;

  for (int chunk = 0; chunk < 16; ++chunk) {
    const uint4* gh = (const uint4*)(cb_hi + chunk * 4096);
    const uint4* gl = (const uint4*)(cb_lo + chunk * 4096);
    __syncthreads();                      // WAR: prior compute done with LDS
    #pragma unroll
    for (int u = tid; u < 512; u += 256) {
      lds_hi[u] = gh[u];
      lds_lo[u] = gl[u];
    }
    __syncthreads();
    if (chunk + 1 < 16) {                 // prefetch next chunk into GL2/WGP$
      __builtin_prefetch((const char*)(cb_hi + (chunk + 1) * 4096) + tid * 32, 0, 0);
      __builtin_prefetch((const char*)(cb_lo + (chunk + 1) * 4096) + tid * 32, 0, 0);
    }

    #pragma unroll
    for (int ct = 0; ct < 4; ++ct) {
      int col = chunk * 64 + ct * 16 + m;     // lanes l and l+16 share column
      float cn = cnorm[col];
      v8f acc;
      #pragma unroll
      for (int r = 0; r < 8; ++r) acc[r] = cn;
      #pragma unroll
      for (int s = 0; s < 2; ++s) {
        v16bf bh = Bh[(ct * 2 + s) * 32 + lane];
        v16bf bl = Bl[(ct * 2 + s) * 32 + lane];
        acc = __builtin_amdgcn_wmma_f32_16x16x32_bf16(false, ah[s].v, false, bh, (short)0, acc, false, false);
        acc = __builtin_amdgcn_wmma_f32_16x16x32_bf16(false, ah[s].v, false, bl, (short)0, acc, false, false);
        acc = __builtin_amdgcn_wmma_f32_16x16x32_bf16(false, al[s].v, false, bh, (short)0, acc, false, false);
      }
      // Running argmin; columns increase monotonically -> '<' keeps first tie.
      #pragma unroll
      for (int r = 0; r < 8; ++r)
        if (acc[r] < rv[r]) { rv[r] = acc[r]; ri[r] = col; }
    }
  }

  // Cross-lane reduce over the 16 columns in each lane-half (masks 1,2,4,8
  // stay within the half).  Ties resolved toward the lowest code index.
  #pragma unroll
  for (int mask = 1; mask <= 8; mask <<= 1) {
    #pragma unroll
    for (int r = 0; r < 8; ++r) {
      float ov = __shfl_xor(rv[r], mask, 32);
      int   oi = __shfl_xor(ri[r], mask, 32);
      if (ov < rv[r] || (ov == rv[r] && oi < ri[r])) { rv[r] = ov; ri[r] = oi; }
    }
  }

  // C-layout: lanes 0-15 hold rows 0..7, lanes 16-31 hold rows 8..15.
  if ((lane & 15) == 0) {
    int rbase = rowbase + h * 8;
    #pragma unroll
    for (int r = 0; r < 8; ++r) {
      out_tok_f[rbase + r] = (float)ri[r];
      atomicAdd(&hist[ri[r]], 1u);
    }
  }
}

// ============================================================================
// Gather zq (= zq_st numerically), write transposed to (B,D,T'), and reduce
// the commit-loss sum in double precision.
__launch_bounds__(256)
__global__ void vq_gather(const float* __restrict__ ze,
                          const float* __restrict__ cb,
                          const float* __restrict__ tok_f,
                          float* __restrict__ zq,
                          double* __restrict__ loss) {
  __shared__ double red[256];
  int n = blockIdx.x * 256 + threadIdx.x;
  int b = n >> 11, t = n & 2047;
  int idx = (int)tok_f[n];
  const float* crow = cb + idx * DD;
  size_t base = (size_t)b * DD * TPP + t;
  double acc = 0.0;
  #pragma unroll 4
  for (int d = 0; d < DD; ++d) {
    float c = crow[d];
    size_t off = base + (size_t)d * TPP;   // consecutive threads -> coalesced
    float z = ze[off];
    zq[off] = c;
    float df = z - c;
    acc += (double)df * (double)df;
  }
  red[threadIdx.x] = acc;
  __syncthreads();
  for (int s = 128; s > 0; s >>= 1) {
    if (threadIdx.x < s) red[threadIdx.x] += red[threadIdx.x + s];
    __syncthreads();
  }
  if (threadIdx.x == 0) atomicAdd(loss, red[0]);   // global_atomic_add_f64
}

// ============================================================================
__global__ void vq_finalize(const unsigned int* __restrict__ hist,
                            const double* __restrict__ loss,
                            float* __restrict__ out) {
  __shared__ float red[1024];
  int k = threadIdx.x;
  float p = (float)hist[k] / (float)NROWS;
  red[k] = p * logf(p + 1e-10f);
  __syncthreads();
  for (int s = 512; s > 0; s >>= 1) {
    if (k < s) red[k] += red[k + s];
    __syncthreads();
  }
  if (k == 0) {
    out[PP_OFF] = expf(-red[0]);
    out[CL_OFF] = 0.25f * (float)(*loss / (double)ZQ_SIZE);
  }
}

// ============================================================================
extern "C" void kernel_launch(void* const* d_in, const int* in_sizes, int n_in,
                              void* d_out, int out_size, void* d_ws, size_t ws_size,
                              hipStream_t stream) {
  const float* ze = (const float*)d_in[0];      // (32,64,2048) f32
  const float* cb = (const float*)d_in[1];      // (1024,64) f32
  float* out = (float*)d_out;

  unsigned char* ws = (unsigned char*)d_ws;
  unsigned int*   hist  = (unsigned int*)ws;
  double*         loss  = (double*)(ws + 4096);
  float*          cnorm = (float*)(ws + 8192);
  unsigned short* cbhi  = (unsigned short*)(ws + 12288);
  unsigned short* cblo  = (unsigned short*)(ws + 12288 + KCODES * DD * sizeof(unsigned short));

  vq_zero    <<<1,   1024, 0, stream>>>(hist, loss);
  vq_prep    <<<4,    256, 0, stream>>>(cb, cbhi, cblo, cnorm);
  vq_argmin  <<<512,  256, 0, stream>>>(ze, cbhi, cblo, cnorm, out + TOK_OFF, hist);
  vq_gather  <<<256,  256, 0, stream>>>(ze, cb, out + TOK_OFF, out, loss);
  vq_finalize<<<1,   1024, 0, stream>>>(hist, loss, out);
}
